// PointTransformerBlock_5609227289187
// MI455X (gfx1250) — matrix operand order repaired and checked by reading, not compiled, not run on hardware
//
#include <hip/hip_runtime.h>
#include <hip/hip_bf16.h>
#include <math.h>

// ---------------------------------------------------------------------------
// Point Transformer block for MI455X (gfx1250, wave32, WMMA).
// All dense GEMMs use v_wmma_f32_16x16x32_f16 (f16 in, f32 accumulate).
// Fragments are fed from LDS with 16-byte ds_load_b128 accesses:
//   A staged row-major [M][Kpad], B staged transposed [N][Kpad].
// ---------------------------------------------------------------------------

typedef __attribute__((ext_vector_type(16))) _Float16 v16h;
typedef __attribute__((ext_vector_type(8)))  _Float16 v8h;
typedef __attribute__((ext_vector_type(8)))  float    v8f;

#define NPTS   4096
#define BATCH  2
#define DIM    256
#define HEADS  8
#define DHEAD  32
#define KNBR   16
#define PEHID  64
#define FFNH   1024

__device__ __forceinline__ float gelu_f(float x) {
    return 0.5f * x * (1.0f + erff(x * 0.70710678118654752f));
}

// ---- WMMA fragment loads from LDS (contiguous K order k = 16*(l>>4) + s, wave32) -------------
// A (16x32 f16, MxK), LDS row-major [row][kstride]:
//   lanes 0-15 -> M=lane, K {0..15}; lanes 16-31 -> K {16..31}.
//   Two contiguous 16B chunks per lane -> 2x ds_load_b128.
__device__ __forceinline__ v16h frag_a_lds(const _Float16* base, int row0,
                                           int koff, int kstride) {
    int l = threadIdx.x & 31;
    int m = l & 15;
    int g = l >> 4;
    const _Float16* p = base + (row0 + m) * kstride + koff + g * 16;
    v8h lo = *(const v8h*)(p);
    v8h hi = *(const v8h*)(p + 8);
    return __builtin_shufflevector(lo, hi, 0, 1, 2, 3, 4, 5, 6, 7,
                                   8, 9, 10, 11, 12, 13, 14, 15);
}

// B (32x16 f16, KxN), LDS staged TRANSPOSED [col][kstride]:
//   lane n = l&15 needs 16 consecutive K starting at (l>>4)*16
//   -> 32 contiguous bytes -> 2x ds_load_b128.
__device__ __forceinline__ v16h frag_bt_lds(const _Float16* base, int col0,
                                            int koff, int kstride) {
    int l = threadIdx.x & 31;
    int n = l & 15;
    int g = l >> 4;
    const _Float16* p = base + (col0 + n) * kstride + koff + g * 16;
    v8h lo = *(const v8h*)(p);
    v8h hi = *(const v8h*)(p + 8);
    return __builtin_shufflevector(lo, hi, 0, 1, 2, 3, 4, 5, 6, 7,
                                   8, 9, 10, 11, 12, 13, 14, 15);
}

__device__ __forceinline__ v8f wmma_f16(v16h a, v16h b, v8f c) {
    return __builtin_amdgcn_wmma_f32_16x16x32_f16(
        /*neg_a=*/false, a, /*neg_b=*/false, b,
        /*c_mod=*/(short)0, c, /*reuse_a=*/false, /*reuse_b=*/false);
}

// ---------------------------------------------------------------------------
// 1) Brute-force KNN (top-16, self excluded), LDS-tiled candidates.
// ---------------------------------------------------------------------------
__global__ __launch_bounds__(256)
void knn_kernel(const float* __restrict__ xyz, int* __restrict__ idx) {
    const int TS = 512;
    __shared__ float sx[TS * 3];
    int p = blockIdx.x * 256 + threadIdx.x;   // global point, one batch/block
    int b = p >> 12;
    int n = p & (NPTS - 1);
    float px = xyz[(size_t)p * 3 + 0];
    float py = xyz[(size_t)p * 3 + 1];
    float pz = xyz[(size_t)p * 3 + 2];
    float bd[KNBR];
    int   bi[KNBR];
#pragma unroll
    for (int i = 0; i < KNBR; ++i) { bd[i] = 3.0e38f; bi[i] = 0; }

    for (int t0 = 0; t0 < NPTS; t0 += TS) {
        __syncthreads();
#pragma unroll
        for (int i = 0; i < 6; ++i)
            sx[threadIdx.x * 6 + i] =
                xyz[((size_t)b * NPTS + t0) * 3 + threadIdx.x * 6 + i];
        __syncthreads();
        for (int c = 0; c < TS; ++c) {
            int cand = t0 + c;
            float dx = px - sx[c * 3 + 0];
            float dy = py - sx[c * 3 + 1];
            float dz = pz - sx[c * 3 + 2];
            float d = dx * dx + dy * dy + dz * dz;
            if (cand != n && d < bd[KNBR - 1]) {
                int j = KNBR - 1;
                while (j > 0 && bd[j - 1] > d) {
                    bd[j] = bd[j - 1];
                    bi[j] = bi[j - 1];
                    --j;
                }
                bd[j] = d;
                bi[j] = cand;
            }
        }
    }
#pragma unroll
    for (int i = 0; i < KNBR; ++i) idx[(size_t)p * KNBR + i] = bi[i];
}

// ---------------------------------------------------------------------------
// 2) LayerNorm over DIM=256, one wave per row; float4 loads, b128 f16 store.
// ---------------------------------------------------------------------------
__global__ __launch_bounds__(256)
void ln_kernel(const float* __restrict__ x, const float* __restrict__ g,
               const float* __restrict__ bta, _Float16* __restrict__ out) {
    int wave = threadIdx.x >> 5;
    int lane = threadIdx.x & 31;
    size_t row = (size_t)blockIdx.x * 8 + wave;
    const float4* xr = (const float4*)(x + row * DIM) + lane * 2;
    float4 a = xr[0];
    float4 b4 = xr[1];
    float v[8] = {a.x, a.y, a.z, a.w, b4.x, b4.y, b4.z, b4.w};
    float s = 0.0f;
#pragma unroll
    for (int i = 0; i < 8; ++i) s += v[i];
#pragma unroll
    for (int o = 16; o > 0; o >>= 1) s += __shfl_xor(s, o, 32);
    float mu = s * (1.0f / DIM);
    float var = 0.0f;
#pragma unroll
    for (int i = 0; i < 8; ++i) { float d = v[i] - mu; var += d * d; }
#pragma unroll
    for (int o = 16; o > 0; o >>= 1) var += __shfl_xor(var, o, 32);
    var *= (1.0f / DIM);
    float r = rsqrtf(var + 1e-5f);
    int c0 = lane * 8;
    v8h o16;
#pragma unroll
    for (int i = 0; i < 8; ++i)
        o16[i] = (_Float16)((v[i] - mu) * r * g[c0 + i] + bta[c0 + i]);
    *(v8h*)(out + row * DIM + c0) = o16;
}

// ---------------------------------------------------------------------------
// 3) Generic WMMA GEMM: C[M,N] = epi(A16[M,K] @ Bw_f32[K,N]).
//    Block tile 64x128, 8 waves (2x4), wave tile 32x32 (4 accum tiles).
// ---------------------------------------------------------------------------
enum { EPI_BIAS = 1, EPI_GELU = 2, EPI_RESID = 4, EPI_F16OUT = 8 };

template <int EPI>
__global__ __launch_bounds__(256)
void wmma_gemm_kernel(const _Float16* __restrict__ A,
                      const float* __restrict__ Bw,
                      const float* __restrict__ bias,
                      const float* __restrict__ resid,
                      float* __restrict__ outF, _Float16* __restrict__ outH,
                      int M, int N, int Kd) {
    const int BM = 64, BN = 128, BK = 32;
    const int AST = 48;                     // padded K-stride (96B rows)
    const int BST = 48;
    __shared__ __align__(16) _Float16 sA[BM * AST];    // 6 KB
    __shared__ __align__(16) _Float16 sBT[BN * BST];   // 12 KB (transposed)
    int tid = threadIdx.x;
    int wave = tid >> 5;
    int wr = wave >> 2;   // 0..1 -> 32-row strip
    int wc = wave & 3;    // 0..3 -> 32-col strip
    int bm = blockIdx.x * BM;
    int bn = blockIdx.y * BN;

    v8f acc[2][2] = {};

    for (int k0 = 0; k0 < Kd; k0 += BK) {
        __syncthreads();
        {   // stage A: 64x32 f16, one 16B vector per thread
            int r = tid >> 2, c = (tid & 3) * 8;
            const _Float16* src = A + (size_t)(bm + r) * Kd + k0 + c;
            *(v8h*)(sA + r * AST + c) = *(const v8h*)src;
            if (k0 + BK < Kd)
                __builtin_prefetch(src + BK, 0, 0);   // global_prefetch_b8
        }
        {   // stage B: 32x128 f32 -> f16, transposed into [col][K]
            int r = tid >> 3, c = (tid & 7) * 16;
            const float* src = Bw + (size_t)(k0 + r) * N + bn + c;
#pragma unroll
            for (int i = 0; i < 16; ++i)
                sBT[(c + i) * BST + r] = (_Float16)src[i];
            if (k0 + BK < Kd)
                __builtin_prefetch(src + (size_t)BK * N, 0, 0);
        }
        __syncthreads();

        v16h af[2], bf[2];
#pragma unroll
        for (int i = 0; i < 2; ++i)
            af[i] = frag_a_lds(sA, wr * 32 + i * 16, 0, AST);
#pragma unroll
        for (int j = 0; j < 2; ++j)
            bf[j] = frag_bt_lds(sBT, wc * 32 + j * 16, 0, BST);
#pragma unroll
        for (int i = 0; i < 2; ++i)
#pragma unroll
            for (int j = 0; j < 2; ++j)
                acc[i][j] = wmma_f16(af[i], bf[j], acc[i][j]);
    }

    // Epilogue: C/D layout -> lane n = l&15, VGPR e -> row (l>>4)*8+e.
    int lane = tid & 31;
    int cn = lane & 15;
    int gg = lane >> 4;
#pragma unroll
    for (int i = 0; i < 2; ++i) {
#pragma unroll
        for (int j = 0; j < 2; ++j) {
            int col = bn + wc * 32 + j * 16 + cn;
#pragma unroll
            for (int e = 0; e < 8; ++e) {
                int row = bm + wr * 32 + i * 16 + gg * 8 + e;
                float c = acc[i][j][e];
                if (EPI & EPI_BIAS)  c += bias[col];
                if (EPI & EPI_GELU)  c = gelu_f(c);
                if (EPI & EPI_RESID) c += resid[(size_t)row * N + col];
                if (EPI & EPI_F16OUT)
                    outH[(size_t)row * N + col] = (_Float16)c;
                else
                    outF[(size_t)row * N + col] = c;
            }
        }
    }
}

// ---------------------------------------------------------------------------
// 4) Fused positional-encoding MLP: rel -> gelu(rel@W1+b1) (VALU) -> @W2+b2
//    via WMMA.  Block = 64 (point,neighbor) rows x 256 cols.
// ---------------------------------------------------------------------------
__global__ __launch_bounds__(256)
void pe_kernel(const float* __restrict__ xyz, const int* __restrict__ idx,
               const float* __restrict__ w1, const float* __restrict__ b1,
               const float* __restrict__ w2, const float* __restrict__ b2,
               _Float16* __restrict__ pe) {
    const int HST = 80;                      // padded K-stride (160B rows)
    const int WST = 80;
    __shared__ __align__(16) _Float16 sH[64 * HST];    // 10 KB
    __shared__ __align__(16) _Float16 sW[DIM * WST];   // 40 KB (transposed W2)
    int tid = threadIdx.x;
    int r0 = blockIdx.x * 64;

    {   // phase 1: hidden = gelu(rel @ W1 + b1), 16 cols per thread
        int lr = tid >> 2;
        int c0 = (tid & 3) * 16;
        int r = r0 + lr;
        int b = r >> 16;                 // N*K = 65536
        int rr = r & 65535;
        int n = rr >> 4;
        int kk = rr & 15;
        int j = idx[((size_t)b * NPTS + n) * KNBR + kk];
        size_t pa = ((size_t)b * NPTS + n) * 3;
        size_t pb = ((size_t)b * NPTS + j) * 3;
        float rx = xyz[pa + 0] - xyz[pb + 0];
        float ry = xyz[pa + 1] - xyz[pb + 1];
        float rz = xyz[pa + 2] - xyz[pb + 2];
#pragma unroll
        for (int c = 0; c < 16; ++c) {
            int col = c0 + c;
            float s = rx * w1[col] + ry * w1[PEHID + col] +
                      rz * w1[2 * PEHID + col] + b1[col];
            sH[lr * HST + col] = (_Float16)gelu_f(s);
        }
    }
    {   // stage W2 (64x256) f32 -> f16, transposed into [col][K]
        int rw = tid >> 2, cw = (tid & 3) * 64;
#pragma unroll
        for (int i = 0; i < 64; ++i)
            sW[(cw + i) * WST + rw] = (_Float16)w2[(size_t)rw * DIM + cw + i];
    }
    __syncthreads();

    int wave = tid >> 5;
    int wr = wave >> 2;   // 0..1 -> 32-row strip
    int wc = wave & 3;    // 0..3 -> 64-col strip
    v8f acc[2][4] = {};
#pragma unroll
    for (int k0 = 0; k0 < PEHID; k0 += 32) {
        v16h af[2], bf[4];
#pragma unroll
        for (int i = 0; i < 2; ++i)
            af[i] = frag_a_lds(sH, wr * 32 + i * 16, k0, HST);
#pragma unroll
        for (int j = 0; j < 4; ++j)
            bf[j] = frag_bt_lds(sW, wc * 64 + j * 16, k0, WST);
#pragma unroll
        for (int i = 0; i < 2; ++i)
#pragma unroll
            for (int j = 0; j < 4; ++j)
                acc[i][j] = wmma_f16(af[i], bf[j], acc[i][j]);
    }

    int lane = tid & 31;
    int cn = lane & 15;
    int gg = lane >> 4;
#pragma unroll
    for (int i = 0; i < 2; ++i) {
#pragma unroll
        for (int j = 0; j < 4; ++j) {
            int col = wc * 64 + j * 16 + cn;
#pragma unroll
            for (int e = 0; e < 8; ++e) {
                int row = r0 + wr * 32 + i * 16 + gg * 8 + e;
                pe[(size_t)row * DIM + col] =
                    (_Float16)(acc[i][j][e] + b2[col]);
            }
        }
    }
}

// ---------------------------------------------------------------------------
// 5) Attention: one wave per (b, n, head); lane = channel (DH == wave32).
// ---------------------------------------------------------------------------
__global__ __launch_bounds__(256)
void attn_kernel(const float* __restrict__ q, const _Float16* __restrict__ k16,
                 const _Float16* __restrict__ v16h_,
                 const _Float16* __restrict__ pe16,
                 const int* __restrict__ idx, _Float16* __restrict__ out16) {
    int wave = threadIdx.x >> 5;
    int lane = threadIdx.x & 31;
    int p = blockIdx.x * 8 + wave;            // (b,n,h) index
    int b = p / (NPTS * HEADS);
    int rem = p - b * NPTS * HEADS;
    int n = rem / HEADS;
    int h = rem - n * HEADS;
    size_t rowq = ((size_t)b * NPTS + n) * DIM + h * DHEAD + lane;
    float qv = q[rowq] * 0.17677669529663687f;   // 1/sqrt(DH)

    float vp[KNBR];
    float lg[KNBR];
#pragma unroll
    for (int kk = 0; kk < KNBR; ++kk) {
        int j = idx[((size_t)b * NPTS + n) * KNBR + kk];
        size_t off = ((size_t)b * NPTS + j) * DIM + h * DHEAD + lane;
        size_t po = (((size_t)b * NPTS + n) * KNBR + kk) * DIM + h * DHEAD + lane;
        float pe = (float)pe16[po];
        float kv = (float)k16[off] + pe;
        vp[kk] = (float)v16h_[off] + pe;
        float s = qv * kv;
#pragma unroll
        for (int o = 16; o > 0; o >>= 1) s += __shfl_xor(s, o, 32);
        lg[kk] = s;
    }
    float mx = lg[0];
#pragma unroll
    for (int kk = 1; kk < KNBR; ++kk) mx = fmaxf(mx, lg[kk]);
    float den = 0.0f;
#pragma unroll
    for (int kk = 0; kk < KNBR; ++kk) { lg[kk] = __expf(lg[kk] - mx); den += lg[kk]; }
    float inv = 1.0f / den;
    float o = 0.0f;
#pragma unroll
    for (int kk = 0; kk < KNBR; ++kk) o += lg[kk] * inv * vp[kk];
    out16[rowq] = (_Float16)o;
}

// ---------------------------------------------------------------------------
// Launch
// ---------------------------------------------------------------------------
extern "C" void kernel_launch(void* const* d_in, const int* in_sizes, int n_in,
                              void* d_out, int out_size, void* d_ws,
                              size_t ws_size, hipStream_t stream) {
    const float* x      = (const float*)d_in[0];
    const float* xyz    = (const float*)d_in[1];
    const float* Wq     = (const float*)d_in[2];
    const float* Wk     = (const float*)d_in[3];
    const float* Wv     = (const float*)d_in[4];
    const float* pe_w1  = (const float*)d_in[5];
    const float* pe_b1  = (const float*)d_in[6];
    const float* pe_w2  = (const float*)d_in[7];
    const float* pe_b2  = (const float*)d_in[8];
    const float* proj_w = (const float*)d_in[9];
    const float* proj_b = (const float*)d_in[10];
    const float* ffn_w1 = (const float*)d_in[11];
    const float* ffn_b1 = (const float*)d_in[12];
    const float* ffn_w2 = (const float*)d_in[13];
    const float* ffn_b2 = (const float*)d_in[14];
    const float* ln1_g  = (const float*)d_in[15];
    const float* ln1_b  = (const float*)d_in[16];
    const float* ln2_g  = (const float*)d_in[17];
    const float* ln2_b  = (const float*)d_in[18];

    const size_t ROWS = (size_t)BATCH * NPTS;           // 8192
    const size_t PROWS = ROWS * KNBR;                   // 131072

    char* ws = (char*)d_ws;
    size_t off = 0;
    auto carve = [&](size_t bytes) {
        void* p = ws + off;
        off += (bytes + 255) & ~(size_t)255;
        return p;
    };
    int*       idx  = (int*)      carve(ROWS * KNBR * sizeof(int));
    _Float16*  h16  = (_Float16*) carve(ROWS * DIM * 2);
    float*     qbuf = (float*)    carve(ROWS * DIM * 4);
    _Float16*  kbuf = (_Float16*) carve(ROWS * DIM * 2);
    _Float16*  vbuf = (_Float16*) carve(ROWS * DIM * 2);
    _Float16*  pe16 = (_Float16*) carve(PROWS * DIM * 2);
    _Float16*  ao16 = (_Float16*) carve(ROWS * DIM * 2);
    float*     x1   = (float*)    carve(ROWS * DIM * 4);
    _Float16*  h2   = (_Float16*) carve(ROWS * DIM * 2);
    _Float16*  ff16 = (_Float16*) carve(ROWS * FFNH * 2);
    (void)ws_size; (void)in_sizes; (void)n_in; (void)out_size;

    float* out = (float*)d_out;

    // 1) KNN
    knn_kernel<<<(int)(ROWS / 256), 256, 0, stream>>>(xyz, idx);

    // 2) LN1 -> h16
    ln_kernel<<<(int)(ROWS / 8), 256, 0, stream>>>(x, ln1_g, ln1_b, h16);

    // 3) q, k, v GEMMs
    dim3 g256((unsigned)(ROWS / 64), DIM / 128);
    wmma_gemm_kernel<0><<<g256, 256, 0, stream>>>(
        h16, Wq, nullptr, nullptr, qbuf, nullptr, (int)ROWS, DIM, DIM);
    wmma_gemm_kernel<EPI_F16OUT><<<g256, 256, 0, stream>>>(
        h16, Wk, nullptr, nullptr, nullptr, kbuf, (int)ROWS, DIM, DIM);
    wmma_gemm_kernel<EPI_F16OUT><<<g256, 256, 0, stream>>>(
        h16, Wv, nullptr, nullptr, nullptr, vbuf, (int)ROWS, DIM, DIM);

    // 4) positional-encoding MLP (fused, WMMA second layer)
    pe_kernel<<<(int)(PROWS / 64), 256, 0, stream>>>(
        xyz, idx, pe_w1, pe_b1, pe_w2, pe_b2, pe16);

    // 5) attention
    attn_kernel<<<(int)(ROWS * HEADS / 8), 256, 0, stream>>>(
        qbuf, kbuf, vbuf, pe16, idx, ao16);

    // 6) proj GEMM + bias + residual(x) -> x1
    wmma_gemm_kernel<EPI_BIAS | EPI_RESID><<<g256, 256, 0, stream>>>(
        ao16, proj_w, proj_b, x, x1, nullptr, (int)ROWS, DIM, DIM);

    // 7) LN2 -> h2
    ln_kernel<<<(int)(ROWS / 8), 256, 0, stream>>>(x1, ln2_g, ln2_b, h2);

    // 8) FFN1: [8192,256]@[256,1024] + bias + gelu -> f16
    dim3 g1024((unsigned)(ROWS / 64), FFNH / 128);
    wmma_gemm_kernel<EPI_BIAS | EPI_GELU | EPI_F16OUT><<<g1024, 256, 0, stream>>>(
        h2, ffn_w1, ffn_b1, nullptr, nullptr, ff16, (int)ROWS, FFNH, DIM);

    // 9) FFN2: [8192,1024]@[1024,256] + bias + residual(x1) -> d_out
    wmma_gemm_kernel<EPI_BIAS | EPI_RESID><<<g256, 256, 0, stream>>>(
        ff16, ffn_w2, ffn_b2, x1, out, nullptr, (int)ROWS, DIM, FFNH);
}
